// TransformerXL_87308095193120
// MI455X (gfx1250) — matrix operand order repaired and checked by reading, not compiled, not run on hardware
//
#include <hip/hip_runtime.h>
#include <hip/hip_bf16.h>
#include <math.h>

// ---------------------------------------------------------------------------
// TransformerXL forward for MI455X (gfx1250, wave32, WMMA).
// GEMMs: f32 -> f16 staged in double-buffered LDS, v_wmma_f32_16x16x32_f16,
// f32 accumulation; 128x64 block tile, 32x32 per-wave tile (4 accumulators),
// software-pipelined k-loop (peeled prologue/epilogue, 1 barrier per step).
// ---------------------------------------------------------------------------

#define L_SEQ   2048
#define D_MOD   1024
#define NHEAD   16
#define DHEAD   64
#define D_FF    4096
#define N_LAYER 12
#define N_VOCAB 32000
#define LN_EPS  1e-5f
#define ATTN_SCALE 0.125f   // 1/sqrt(64)

typedef __attribute__((ext_vector_type(16))) _Float16 v16h;
typedef __attribute__((ext_vector_type(8)))  float    v8f;

#define BM   128  // block tile M
#define BN   64   // block tile N
#define TK   32   // K step (one WMMA K)
#define ASTR 36   // LDS row stride (halves), padded & even
#define BSTR 36

union Frag { v16h h; unsigned int u[8]; };

__device__ __forceinline__ void cvt_store8(_Float16* d, float4 x0, float4 x1)
{
    d[0]=(_Float16)x0.x; d[1]=(_Float16)x0.y; d[2]=(_Float16)x0.z; d[3]=(_Float16)x0.w;
    d[4]=(_Float16)x1.x; d[5]=(_Float16)x1.y; d[6]=(_Float16)x1.z; d[7]=(_Float16)x1.w;
}

// A tile: BM x TK, 16 contiguous floats per thread (4x b128 loads).
__device__ __forceinline__ void stage_a(const float* __restrict__ A, int lda,
                                        int m0, int k0, int tid, _Float16* dst)
{
    int row = tid >> 1;
    int col = (tid & 1) * 16;
    const float4* ap = (const float4*)(A + (size_t)(m0 + row) * lda + (k0 + col));
    float4 x0 = ap[0], x1 = ap[1], x2 = ap[2], x3 = ap[3];
    _Float16* d = dst + row * ASTR + col;
    cvt_store8(d,     x0, x1);
    cvt_store8(d + 8, x2, x3);
}

// B tile (NT: B stored (N,K)): LDS layout [n][k], contiguous k loads.
__device__ __forceinline__ void stage_b_nt(const float* __restrict__ B, int ldb,
                                           int n0, int k0, int tid, _Float16* dst)
{
    int row = tid >> 2;
    int col = (tid & 3) * 8;
    const float4* bp = (const float4*)(B + (size_t)(n0 + row) * ldb + (k0 + col));
    float4 x0 = bp[0], x1 = bp[1];
    cvt_store8(dst + row * BSTR + col, x0, x1);
}

// B tile (NN: B stored (K,N)): contiguous-n global loads, scatter into [n][k].
__device__ __forceinline__ void stage_b_nn(const float* __restrict__ B, int ldb,
                                           int n0, int k0, int tid, _Float16* dst)
{
    int kr = tid >> 3;
    int nc = (tid & 7) * 8;
    const float4* bp = (const float4*)(B + (size_t)(k0 + kr) * ldb + (n0 + nc));
    float4 x0 = bp[0], x1 = bp[1];
    dst[(nc + 0) * BSTR + kr] = (_Float16)x0.x;
    dst[(nc + 1) * BSTR + kr] = (_Float16)x0.y;
    dst[(nc + 2) * BSTR + kr] = (_Float16)x0.z;
    dst[(nc + 3) * BSTR + kr] = (_Float16)x0.w;
    dst[(nc + 4) * BSTR + kr] = (_Float16)x1.x;
    dst[(nc + 5) * BSTR + kr] = (_Float16)x1.y;
    dst[(nc + 6) * BSTR + kr] = (_Float16)x1.z;
    dst[(nc + 7) * BSTR + kr] = (_Float16)x1.w;
}

// ISA 7.12.2 wave32 fragment layouts.
__device__ __forceinline__ v16h load_afrag(const _Float16* Al, int rowbase, int hi)
{
    Frag f;
    const int r = rowbase * ASTR;
    #pragma unroll
    for (int t = 0; t < 4; ++t) {
        f.u[t]     = *(const unsigned int*)&Al[r + hi * 8 + 2 * t];        // K 0..7 / 8..15
        f.u[t + 4] = *(const unsigned int*)&Al[r + 16 + hi * 8 + 2 * t];   // K 16..23 / 24..31
    }
    return f.h;
}

__device__ __forceinline__ v16h load_bfrag(const _Float16* Bl, int colbase, int hi)
{
    Frag f;
    const int r = colbase * BSTR;
    #pragma unroll
    for (int t = 0; t < 8; ++t)
        f.u[t] = *(const unsigned int*)&Bl[r + (t >> 2) * 16 + hi * 8 + 2 * (t & 3)]; // K 0..7,16..23 / 8..15,24..31
    return f.h;
}

#define WMMA_F16(a, b, c) \
    __builtin_amdgcn_wmma_f32_16x16x32_f16(false, (a), false, (b), (short)0, (c), false, false)

// C[M,N] = op( A @ B  (+ A2 @ B2 if DUAL)  + bias? ), exact 128/64/32 tiling.
// TRANSB=1: B stored (N,K) row-major (NT).  TRANSB=0: B stored (K,N) (NN).
// DUAL second product is always NT (AC+BD score fusion). Requires K >= 2*TK.
template<int TRANSB, int DUAL>
__global__ __launch_bounds__(256)
void gemm_wmma(const float* __restrict__ A,  int lda,
               const float* __restrict__ B,  int ldb,
               const float* __restrict__ A2, int lda2,
               const float* __restrict__ B2, int ldb2,
               float* __restrict__ C, int ldc, int K,
               const float* __restrict__ bias, int relu)
{
    __shared__ _Float16 Al [2][BM * ASTR];
    __shared__ _Float16 Bl [2][BN * BSTR];
    __shared__ _Float16 Al2[2][DUAL ? BM * ASTR : 2];
    __shared__ _Float16 Bl2[2][DUAL ? BN * BSTR : 2];

    const int tid  = threadIdx.x;
    const int m0   = blockIdx.y * BM;
    const int n0   = blockIdx.x * BN;
    const int lane = tid & 31;
    const int wv   = tid >> 5;           // 0..7
    const int mw   = (wv & 3) * 32;      // wave M offset (4 waves x 32 rows)
    const int nw   = (wv >> 2) * 32;     // wave N offset (2 waves x 32 cols)
    const int hl   = lane & 15;
    const int hi   = lane >> 4;

    v8f acc00 = {}, acc01 = {}, acc10 = {}, acc11 = {};

    auto stage_all = [&](int kt, int pb) {
        const int k0 = kt * TK;
        stage_a(A, lda, m0, k0, tid, Al[pb]);
        if constexpr (TRANSB) stage_b_nt(B, ldb, n0, k0, tid, Bl[pb]);
        else                  stage_b_nn(B, ldb, n0, k0, tid, Bl[pb]);
        if constexpr (DUAL) {
            stage_a(A2, lda2, m0, k0, tid, Al2[pb]);
            stage_b_nt(B2, ldb2, n0, k0, tid, Bl2[pb]);
        }
        // Speculative prefetch one K-tile ahead of the staged tile; per ISA,
        // OOB speculative prefetches are dropped, so no guard branch needed.
        __builtin_prefetch(A + (size_t)(m0 + (tid >> 1)) * lda + (k0 + TK + (tid & 1) * 16), 0, 3);
        if constexpr (TRANSB)
            __builtin_prefetch(B + (size_t)(n0 + (tid >> 2)) * ldb + (k0 + TK + (tid & 3) * 8), 0, 3);
        else
            __builtin_prefetch(B + (size_t)(k0 + TK + (tid >> 3)) * ldb + (n0 + (tid & 7) * 8), 0, 3);
    };

    auto compute = [&](int pb) {
        v16h a0 = load_afrag(Al[pb], mw + hl,      hi);
        v16h a1 = load_afrag(Al[pb], mw + 16 + hl, hi);
        v16h b0 = load_bfrag(Bl[pb], nw + hl,      hi);
        v16h b1 = load_bfrag(Bl[pb], nw + 16 + hl, hi);
        acc00 = WMMA_F16(a0, b0, acc00);
        acc01 = WMMA_F16(a0, b1, acc01);
        acc10 = WMMA_F16(a1, b0, acc10);
        acc11 = WMMA_F16(a1, b1, acc11);
        if constexpr (DUAL) {
            v16h c0 = load_afrag(Al2[pb], mw + hl,      hi);
            v16h c1 = load_afrag(Al2[pb], mw + 16 + hl, hi);
            v16h d0 = load_bfrag(Bl2[pb], nw + hl,      hi);
            v16h d1 = load_bfrag(Bl2[pb], nw + 16 + hl, hi);
            acc00 = WMMA_F16(c0, d0, acc00);
            acc01 = WMMA_F16(c0, d1, acc01);
            acc10 = WMMA_F16(c1, d0, acc10);
            acc11 = WMMA_F16(c1, d1, acc11);
        }
    };

    // Software pipeline, peeled: stage 0; {stage kt+1 | compute kt}; compute last.
    const int nk = K / TK;               // always >= 2 in this network
    stage_all(0, 0);
    __syncthreads();
    int pb = 0;
    for (int kt = 0; kt < nk - 1; ++kt) {
        stage_all(kt + 1, pb ^ 1);       // fill other buffer (unconditional)
        compute(pb);                     // consume current buffer
        __syncthreads();                 // single barrier per step
        pb ^= 1;
    }
    compute(pb);                         // drain final tile (no barrier needed)

    // Epilogue: C 16x16 f32 lane layout (8 VGPRs per fragment).
    #pragma unroll
    for (int rr = 0; rr < 8; ++rr) {
        int r0 = m0 + mw + hi * 8 + rr;
        int r1 = r0 + 16;
        int c0 = n0 + nw + hl;
        int c1 = c0 + 16;
        float v00 = acc00[rr], v01 = acc01[rr];
        float v10 = acc10[rr], v11 = acc11[rr];
        if (bias) {
            float bb0 = bias[c0], bb1 = bias[c1];
            v00 += bb0; v01 += bb1; v10 += bb0; v11 += bb1;
        }
        if (relu) {
            v00 = fmaxf(v00, 0.0f); v01 = fmaxf(v01, 0.0f);
            v10 = fmaxf(v10, 0.0f); v11 = fmaxf(v11, 0.0f);
        }
        C[(size_t)r0 * ldc + c0] = v00;
        C[(size_t)r0 * ldc + c1] = v01;
        C[(size_t)r1 * ldc + c0] = v10;
        C[(size_t)r1 * ldc + c1] = v11;
    }
}

// ---------------------------------------------------------------------------
// Elementwise / reduction kernels
// ---------------------------------------------------------------------------

__global__ void embed_kernel(const int* __restrict__ X,
                             const float* __restrict__ emb,
                             float* __restrict__ hid)
{
    int row = blockIdx.x;
    const float4* s = (const float4*)(emb + (size_t)X[row] * D_MOD);
    float4*       d = (float4*)(hid + (size_t)row * D_MOD);
    d[threadIdx.x] = s[threadIdx.x];   // 256 threads * 4 floats = 1024
}

__global__ void pe_kernel(float* __restrict__ pe)
{
    int k = blockIdx.x;
    float pos = (float)(L_SEQ - 1 - k);
    for (int j = threadIdx.x; j < D_MOD; j += 256) {
        int   i    = (j < D_MOD / 2) ? j : j - D_MOD / 2;
        float freq = expf(-((float)(2 * i) / (float)D_MOD) * 9.210340371976184f); // ln(1e4)
        float ang  = pos * freq;
        pe[(size_t)k * D_MOD + j] = (j < D_MOD / 2) ? sinf(ang) : cosf(ang);
    }
}

__global__ void adduv_kernel(const float* __restrict__ qkv,
                             const float* __restrict__ u,
                             const float* __restrict__ v,
                             float* __restrict__ Qu,
                             float* __restrict__ Qv)
{
    int row = blockIdx.x;
    for (int c = threadIdx.x; c < D_MOD; c += 256) {
        float q = qkv[(size_t)row * (3 * D_MOD) + c];   // Q section
        Qu[(size_t)row * D_MOD + c] = q + u[c];
        Qv[(size_t)row * D_MOD + c] = q + v[c];
    }
}

// causal masked softmax over one row of S (2048 cols); applies ATTN_SCALE.
__global__ void softmax_kernel(float* __restrict__ S)
{
    __shared__ float red[256];
    int q = blockIdx.x;
    int tid = threadIdx.x;
    float* row = S + (size_t)q * L_SEQ;
    int valid = q + 1;

    float mx = -3.402823e38f;
    for (int k = tid; k < valid; k += 256) mx = fmaxf(mx, row[k] * ATTN_SCALE);
    red[tid] = mx; __syncthreads();
    for (int s = 128; s > 0; s >>= 1) {
        if (tid < s) red[tid] = fmaxf(red[tid], red[tid + s]);
        __syncthreads();
    }
    mx = red[0]; __syncthreads();

    float sum = 0.0f;
    for (int k = tid; k < valid; k += 256) {
        float e = expf(row[k] * ATTN_SCALE - mx);
        row[k] = e;
        sum += e;
    }
    red[tid] = sum; __syncthreads();
    for (int s = 128; s > 0; s >>= 1) {
        if (tid < s) red[tid] += red[tid + s];
        __syncthreads();
    }
    float inv = 1.0f / red[0];

    for (int k = tid; k < valid; k += 256) row[k] *= inv;
    for (int k = valid + tid; k < L_SEQ; k += 256) row[k] = 0.0f;
}

// hid = LayerNorm(hid + delta), elementwise_affine=False, biased variance.
__global__ void ln_kernel(float* __restrict__ hid, const float* __restrict__ delta)
{
    __shared__ float r1[256];
    __shared__ float r2[256];
    int row = blockIdx.x;
    int tid = threadIdx.x;
    float x[4];
    float s = 0.0f, s2 = 0.0f;
    #pragma unroll
    for (int i = 0; i < 4; ++i) {
        int c = tid * 4 + i;
        float xx = hid[(size_t)row * D_MOD + c] + delta[(size_t)row * D_MOD + c];
        x[i] = xx; s += xx; s2 += xx * xx;
    }
    r1[tid] = s; r2[tid] = s2; __syncthreads();
    for (int st = 128; st > 0; st >>= 1) {
        if (tid < st) { r1[tid] += r1[tid + st]; r2[tid] += r2[tid + st]; }
        __syncthreads();
    }
    float mean = r1[0] * (1.0f / D_MOD);
    float var  = r2[0] * (1.0f / D_MOD) - mean * mean;
    float rstd = rsqrtf(var + LN_EPS);
    #pragma unroll
    for (int i = 0; i < 4; ++i) {
        int c = tid * 4 + i;
        hid[(size_t)row * D_MOD + c] = (x[i] - mean) * rstd;
    }
}

// ---------------------------------------------------------------------------
// Host orchestration
// ---------------------------------------------------------------------------

template<int TRANSB, int DUAL>
static inline void launch_gemm(hipStream_t s,
                               const float* A, int lda,
                               const float* B, int ldb,
                               const float* A2, int lda2,
                               const float* B2, int ldb2,
                               float* C, int ldc, int M, int N, int K,
                               const float* bias, int relu)
{
    dim3 grid(N / BN, M / BM);
    gemm_wmma<TRANSB, DUAL><<<grid, dim3(256), 0, s>>>(A, lda, B, ldb,
                                                       A2, lda2, B2, ldb2,
                                                       C, ldc, K, bias, relu);
}

extern "C" void kernel_launch(void* const* d_in, const int* in_sizes, int n_in,
                              void* d_out, int out_size, void* d_ws, size_t ws_size,
                              hipStream_t stream)
{
    (void)in_sizes; (void)n_in; (void)out_size; (void)ws_size;

    const int*   X    = (const int*)  d_in[0];
    // d_in[1] = Y (unused: y_len == q_len)
    const float* emb  = (const float*)d_in[2];
    const float* u    = (const float*)d_in[3];
    const float* v    = (const float*)d_in[4];
    const float* Wqkv = (const float*)d_in[5];
    const float* Wpe  = (const float*)d_in[6];
    const float* Wout = (const float*)d_in[7];
    const float* W1   = (const float*)d_in[8];
    const float* b1   = (const float*)d_in[9];
    const float* W2   = (const float*)d_in[10];
    const float* b2   = (const float*)d_in[11];
    const float* Wc   = (const float*)d_in[12];
    const float* bc   = (const float*)d_in[13];
    float* out = (float*)d_out;

    const size_t LD = (size_t)L_SEQ * D_MOD;
    float* ws  = (float*)d_ws;
    float* hid = ws;  ws += LD;                        // (L, D)
    float* pe  = ws;  ws += LD;                        // (L, D)
    float* r   = ws;  ws += LD;                        // (L, D) per-layer
    float* qkv = ws;  ws += (size_t)L_SEQ * 3 * D_MOD; // (L, 3D)
    float* Qu  = ws;  ws += LD;                        // (L, D)
    float* Qv  = ws;  ws += LD;                        // (L, D)
    float* S   = ws;  ws += (size_t)L_SEQ * L_SEQ;     // (L, L), reused per head
    float* av  = ws;  ws += LD;                        // (L, D)
    float* tmp = ws;  ws += LD;                        // (L, D)
    float* ff  = ws;  ws += (size_t)L_SEQ * D_FF;      // (L, DI)

    embed_kernel<<<L_SEQ, 256, 0, stream>>>(X, emb, hid);
    pe_kernel<<<L_SEQ, 256, 0, stream>>>(pe);

    for (int l = 0; l < N_LAYER; ++l) {
        const float* Wqkvl = Wqkv + (size_t)l * (3 * NHEAD * DHEAD) * D_MOD;
        const float* Wpel  = Wpe  + (size_t)l * (NHEAD * DHEAD) * D_MOD;
        const float* Woutl = Wout + (size_t)l * D_MOD * (NHEAD * DHEAD);
        const float* W1l   = W1   + (size_t)l * D_FF * D_MOD;
        const float* b1l   = b1   + (size_t)l * D_FF;
        const float* W2l   = W2   + (size_t)l * D_MOD * D_FF;
        const float* b2l   = b2   + (size_t)l * D_MOD;

        // qkv = hid @ Wqkv.T    (L x 3D)
        launch_gemm<1, 0>(stream, hid, D_MOD, Wqkvl, D_MOD, nullptr, 0, nullptr, 0,
                          qkv, 3 * D_MOD, L_SEQ, 3 * D_MOD, D_MOD, nullptr, 0);
        // r = pe @ Wpe.T        (L x D)
        launch_gemm<1, 0>(stream, pe, D_MOD, Wpel, D_MOD, nullptr, 0, nullptr, 0,
                          r, D_MOD, L_SEQ, D_MOD, D_MOD, nullptr, 0);
        // Qu = Q + u ; Qv = Q + v
        adduv_kernel<<<L_SEQ, 256, 0, stream>>>(qkv, u, v, Qu, Qv);

        for (int h = 0; h < NHEAD; ++h) {
            const float* Kh = qkv + D_MOD + h * DHEAD;       // (L, DH), ldb 3D
            const float* Vh = qkv + 2 * D_MOD + h * DHEAD;   // (L, DH), ldb 3D
            const float* rh = r + h * DHEAD;                 // (L, DH), ldb D
            // S = (Q+u) @ K.T + (Q+v) @ r.T   (fused dual NT GEMM, unshifted BD)
            launch_gemm<1, 1>(stream, Qu + h * DHEAD, D_MOD, Kh, 3 * D_MOD,
                              Qv + h * DHEAD, D_MOD, rh, D_MOD,
                              S, L_SEQ, L_SEQ, L_SEQ, DHEAD, nullptr, 0);
            // causal softmax with scale
            softmax_kernel<<<L_SEQ, 256, 0, stream>>>(S);
            // av[:, h] = P @ V   (NN GEMM, N = 64)
            launch_gemm<0, 0>(stream, S, L_SEQ, Vh, 3 * D_MOD, nullptr, 0, nullptr, 0,
                              av + h * DHEAD, D_MOD, L_SEQ, DHEAD, L_SEQ, nullptr, 0);
        }

        // tmp = av @ Wout.T ; hid = LN(hid + tmp)
        launch_gemm<1, 0>(stream, av, D_MOD, Woutl, NHEAD * DHEAD, nullptr, 0, nullptr, 0,
                          tmp, D_MOD, L_SEQ, D_MOD, NHEAD * DHEAD, nullptr, 0);
        ln_kernel<<<L_SEQ, 256, 0, stream>>>(hid, tmp);

        // ff = relu(hid @ W1.T + b1) ; tmp = ff @ W2.T + b2 ; hid = LN(hid + tmp)
        launch_gemm<1, 0>(stream, hid, D_MOD, W1l, D_MOD, nullptr, 0, nullptr, 0,
                          ff, D_FF, L_SEQ, D_FF, D_MOD, b1l, 1);
        launch_gemm<1, 0>(stream, ff, D_FF, W2l, D_FF, nullptr, 0, nullptr, 0,
                          tmp, D_MOD, L_SEQ, D_MOD, D_FF, b2l, 0);
        ln_kernel<<<L_SEQ, 256, 0, stream>>>(hid, tmp);
    }

    // logits = hid @ Wc.T + bc   (L x V), written straight to d_out
    launch_gemm<1, 0>(stream, hid, D_MOD, Wc, D_MOD, nullptr, 0, nullptr, 0,
                      out, N_VOCAB, L_SEQ, N_VOCAB, D_MOD, bc, 0);
}
